// LeviCivitaKANLayer_54820962566354
// MI455X (gfx1250) — compile-verified, matrix operand order checked by reading, not run on hardware
//
#include <hip/hip_runtime.h>
#include <hip/hip_bf16.h>

#define BB 4
#define QQ 256
#define DD 256
#define OO 256

typedef __attribute__((ext_vector_type(16))) __bf16 v16bf;
typedef __attribute__((ext_vector_type(8)))  float  v8f;
typedef __attribute__((ext_vector_type(4)))  unsigned int u32x4;
typedef __attribute__((ext_vector_type(8)))  int i32x8;
typedef __attribute__((ext_vector_type(4)))  int i32x4;

__device__ __forceinline__ __bf16 f2bf(float f) { return (__bf16)f; }

// K index held by VGPR-pair v (0..7) for a 16-bit 16x32 A / 32x16 B operand,
// per CDNA5 ISA 7.12.2 (lane half selects +8).
__device__ __forceinline__ int kmap(int v, int hf) {
    return (v < 4 ? 2 * v : 16 + 2 * (v - 4)) + hf * 8;
}

// ---------------------------------------------------------------------------
// TDM: DMA a 32x32 f32 tile (rows strided 65536 floats = one lc_w "i" row)
// from global memory into LDS, with hardware padding of +2 dwords every
// 32 dwords -> LDS row stride 34 floats (bank-conflict-free, 8B aligned).
// D# group0/group1 per CDNA5 ISA 8.3/8.4; groups 2/3/iteration zero (2D).
// 6-arg builtin variant (therock-10.0 headers).
// ---------------------------------------------------------------------------
__device__ __forceinline__ void tdm_load_tile(unsigned lds_addr,
                                              unsigned long long ga) {
    unsigned lo = (unsigned)ga;
    unsigned hi = (unsigned)(ga >> 32);
    lds_addr = (unsigned)__builtin_amdgcn_readfirstlane((int)lds_addr);
    lo = (unsigned)__builtin_amdgcn_readfirstlane((int)lo);
    hi = (unsigned)__builtin_amdgcn_readfirstlane((int)hi);
    u32x4 g0;
    g0[0] = 1u;                                   // count=1 (valid user D#)
    g0[1] = lds_addr;                             // LDS byte address
    g0[2] = lo;                                   // global_addr[31:0]
    g0[3] = (hi & 0x01FFFFFFu) | 0x80000000u;     // global_addr[56:32] | type=2
    i32x8 g1;
    g1[0] = 0x03120000;   // data_size=4B | pad_en | pad_interval=32dw | pad=2dw
    g1[1] = 0x01000000;   // tensor_dim0 = 256 (<<16)
    g1[2] = 0;            // tensor_dim0 hi / tensor_dim1 lo
    g1[3] = 0x00200001;   // tile_dim0 = 32 (<<16) | tensor_dim1 hi (=65536)
    g1[4] = 32;           // tile_dim1 = 32, tile_dim2 = 0
    g1[5] = 0x00010000;   // tensor_dim0_stride = 65536 (lo)
    g1[6] = 0;
    g1[7] = 0;
    i32x4 z4 = {0, 0, 0, 0};
    i32x8 z8 = {0, 0, 0, 0, 0, 0, 0, 0};
    __builtin_amdgcn_tensor_load_to_lds(g0, g1, z4, z4, z8, 0);
}

// ---------------------------------------------------------------------------
// Kernel 1: rdenom[b,j,k] = 1 / (1 - sum_q x[b,q,j]^2 * x[b,q,k]^2)
// ---------------------------------------------------------------------------
__global__ void gram_rdenom_kernel(const float* __restrict__ x,
                                   float* __restrict__ rdenom) {
    const int tk = threadIdx.x, tj = threadIdx.y;
    const int j0 = blockIdx.x * 16, k0 = blockIdx.y * 16, b = blockIdx.z;
    __shared__ float Aj[16][17];
    __shared__ float Ak[16][17];
    float acc = 0.f;
    const float* xb = x + (size_t)b * QQ * DD;
    for (int q0 = 0; q0 < QQ; q0 += 16) {
        float a = xb[(q0 + tj) * DD + j0 + tk];
        float c = xb[(q0 + tj) * DD + k0 + tk];
        Aj[tj][tk] = a * a;
        Ak[tj][tk] = c * c;
        __syncthreads();
#pragma unroll
        for (int qq = 0; qq < 16; ++qq)
            acc += Aj[qq][tj] * Ak[qq][tk];
        __syncthreads();
    }
    rdenom[(size_t)b * DD * DD + (j0 + tj) * DD + (k0 + tk)] = 1.0f / (1.0f - acc);
}

// ---------------------------------------------------------------------------
// Kernel 2: y[b,q,i] = bias[i] + sum_d x[b,q,d] * w1[i,d]
// ---------------------------------------------------------------------------
__global__ void base_kernel(const float* __restrict__ x,
                            const float* __restrict__ w1,
                            const float* __restrict__ bias,
                            float* __restrict__ y) {
    const int bq = blockIdx.x;        // b*QQ + q
    const int i = threadIdx.x;
    __shared__ float xrow[DD];
    xrow[i] = x[(size_t)bq * DD + i];
    __syncthreads();
    const float4* wv = (const float4*)(w1 + (size_t)i * DD);
    const float4* xv = (const float4*)xrow;
    float acc = bias[i];
#pragma unroll 4
    for (int d = 0; d < DD / 4; ++d) {
        float4 w = wv[d], xx = xv[d];
        acc += w.x * xx.x + w.y * xx.y + w.z * xx.z + w.w * xx.w;
    }
    y[(size_t)bq * OO + i] = acc;
}

// ---------------------------------------------------------------------------
// Kernel 3: split-K batched GEMM with on-the-fly A generation and TDM-fed,
// double-buffered B staging.
//   lc_out[b,q,i] += sum_{j in chunk,k} (x_j*x_k) * (W[i,j,k]*rdenom*(j!=k))
// Workgroup: 8 waves, tile = 64 q x 256 i, one 16-j chunk of the K space.
// Wave: 4 q-subtiles x 2 i-subtiles -> 8 x v_wmma_f32_16x16x32_bf16 / K-step.
// Current-buffer selection is pure offset arithmetic on the LDS base so all
// staging reads lower to ds_load (not flat), keeping the loop on DScnt only.
// ---------------------------------------------------------------------------
#define BUF_F (32 * 34)     // padded floats per staged 32x32 tile

__global__ void __launch_bounds__(256, 1)
lc_gemm_kernel(const float* __restrict__ x,
               const float* __restrict__ lcw,
               const float* __restrict__ rdenom,
               float* __restrict__ y) {
    const int tid  = threadIdx.x;
    const int lane = tid & 31;
    const int m    = lane & 15;
    const int hf   = lane >> 4;
    const int qblock = blockIdx.x;    // 0..3   (64 q each)
    const int jchunk = blockIdx.y;    // 0..15  (16 j each)
    const int b      = blockIdx.z;
    // Uniform wave index in an SGPR so the whole TDM address chain is scalar.
    const int wave = __builtin_amdgcn_readfirstlane(tid >> 5);

    extern __shared__ char smem_raw[];
    float* xs = (float*)smem_raw;       // [64][258]  x rows (padded, 8B aligned)
    float* rd = xs + 64 * 258;          // [16][256]  rdenom slice
    float* bs = rd + 16 * 256;          // [8][2][BUF_F] per-wave double buffers

    // Cooperative loads of x rows and the rdenom slice for this (b, jchunk).
    const float* xb = x + ((size_t)b * QQ + qblock * 64) * DD;
    for (int idx = tid; idx < 64 * DD; idx += 256) {
        int q = idx >> 8, k = idx & 255;
        xs[q * 258 + k] = xb[q * DD + k];
    }
    const float* rdb = rdenom + ((size_t)b * DD + jchunk * 16) * DD;
    for (int idx = tid; idx < 16 * DD; idx += 256)
        rd[idx] = rdb[idx];

    const int i0 = wave * 32;
    float* bsw = bs + wave * 2 * BUF_F;          // this wave's double buffer
    const unsigned buf_addr0 = (unsigned)(uintptr_t)bsw;
    const unsigned buf_addr1 = buf_addr0 + BUF_F * 4u;
    const unsigned long long gbase =
        (unsigned long long)(uintptr_t)lcw + (unsigned long long)i0 * (DD * DD * 4);

    // Prime the pipeline: step 0 tile (jloc=0, kc=0).
    tdm_load_tile(buf_addr0, gbase + (unsigned long long)(jchunk * 16 * DD) * 4u);
    __syncthreads();   // xs/rd visible to all waves

    v8f acc[4][2];
    const v8f vzero = {0.f, 0.f, 0.f, 0.f, 0.f, 0.f, 0.f, 0.f};
#pragma unroll
    for (int s = 0; s < 4; ++s)
#pragma unroll
        for (int t = 0; t < 2; ++t)
            acc[s][t] = vzero;

    const int NSTEP = 16 * 8;           // 16 jloc x 8 k-chunks of 32
    for (int s = 0; s < NSTEP; ++s) {
        const int jloc = s >> 3;
        const int kc   = (s & 7) << 5;
        const int jj   = jchunk * 16 + jloc;

        if (s + 1 < NSTEP) {
            const int s2 = s + 1;
            const int jj2 = jchunk * 16 + (s2 >> 3);
            const int kc2 = (s2 & 7) << 5;
            tdm_load_tile((s2 & 1) ? buf_addr1 : buf_addr0,
                          gbase + (unsigned long long)((jj2 * DD + kc2) * 4u));
            __builtin_amdgcn_s_wait_tensorcnt(1);
        } else {
            __builtin_amdgcn_s_wait_tensorcnt(0);
        }
        asm volatile("" ::: "memory");

        // Offset arithmetic on the LDS base keeps these reads on the DS path.
        const float* cur = bsw + (s & 1) * BUF_F;
        const float* rrow = rd + jloc * 256 + kc;
        const int dj = jj - kc;         // diagonal hits only when 0 <= dj < 32

        // ---- B fragments (32x16 bf16, two i-subtiles): W * rdenom, diag=0
        v16bf bfrag[2];
#pragma unroll
        for (int v = 0; v < 8; ++v) {
            const int kk = kmap(v, hf);
            const float2 r = *(const float2*)(rrow + kk);
#pragma unroll
            for (int t = 0; t < 2; ++t) {
                const float2 p = *(const float2*)(cur + (t * 16 + m) * 34 + kk);
                float f0 = (kk == dj)     ? 0.f : p.x * r.x;
                float f1 = (kk + 1 == dj) ? 0.f : p.y * r.y;
                bfrag[t][2 * v]     = f2bf(f0);
                bfrag[t][2 * v + 1] = f2bf(f1);
            }
        }

        // ---- per q-subtile: build A fragment (x_j * x_k), fire 2 WMMAs
#pragma unroll
        for (int ss = 0; ss < 4; ++ss) {
            const float* xq = xs + (ss * 16 + m) * 258;
            const float xj = xq[jj];
            v16bf afrag;
#pragma unroll
            for (int v = 0; v < 8; ++v) {
                const int kk = kmap(v, hf);
                const float2 xk = *(const float2*)(xq + kc + kk);
                afrag[2 * v]     = f2bf(xj * xk.x);
                afrag[2 * v + 1] = f2bf(xj * xk.y);
            }
            acc[ss][0] = __builtin_amdgcn_wmma_f32_16x16x32_bf16(
                false, afrag, false, bfrag[0], (short)0, acc[ss][0], false, false);
            acc[ss][1] = __builtin_amdgcn_wmma_f32_16x16x32_bf16(
                false, afrag, false, bfrag[1], (short)0, acc[ss][1], false, false);
        }

        asm volatile("" ::: "memory");  // keep ds reads before next TDM issue
    }

    // ---- split-K reduction: atomically accumulate into y (holds base term)
    float* yb = y + ((size_t)b * QQ + qblock * 64) * OO;
#pragma unroll
    for (int ss = 0; ss < 4; ++ss)
#pragma unroll
        for (int t = 0; t < 2; ++t)
#pragma unroll
            for (int r = 0; r < 8; ++r) {
                const int q = ss * 16 + r + hf * 8;
                const int i = i0 + t * 16 + m;
                unsafeAtomicAdd(&yb[q * OO + i], acc[ss][t][r]);
            }
}

// ---------------------------------------------------------------------------
// Kernel 4: BatchNorm1d over channels = q, stats over (b, o), biased var.
// ---------------------------------------------------------------------------
__global__ void bn_kernel(const float* __restrict__ y,
                          const float* __restrict__ gamma,
                          const float* __restrict__ beta,
                          float* __restrict__ out) {
    const int q = blockIdx.x;
    const int o = threadIdx.x;
    __shared__ float s1[256];
    __shared__ float s2[256];
    float v[BB];
    float s = 0.f, sq = 0.f;
#pragma unroll
    for (int b = 0; b < BB; ++b) {
        v[b] = y[((size_t)b * QQ + q) * OO + o];
        s += v[b];
        sq += v[b] * v[b];
    }
    s1[o] = s;
    s2[o] = sq;
    __syncthreads();
    for (int stride = 128; stride > 0; stride >>= 1) {
        if (o < stride) {
            s1[o] += s1[o + stride];
            s2[o] += s2[o + stride];
        }
        __syncthreads();
    }
    const float mean = s1[0] * (1.0f / 1024.0f);
    const float var  = s2[0] * (1.0f / 1024.0f) - mean * mean;
    const float rstd = rsqrtf(var + 1e-5f);
    const float g = gamma[q], be = beta[q];
#pragma unroll
    for (int b = 0; b < BB; ++b)
        out[((size_t)b * QQ + q) * OO + o] = g * ((v[b] - mean) * rstd) + be;
}

// ---------------------------------------------------------------------------
extern "C" void kernel_launch(void* const* d_in, const int* in_sizes, int n_in,
                              void* d_out, int out_size, void* d_ws, size_t ws_size,
                              hipStream_t stream) {
    const float* x     = (const float*)d_in[0];
    const float* lcw   = (const float*)d_in[1];
    const float* w1    = (const float*)d_in[2];
    const float* bias  = (const float*)d_in[3];
    const float* gamma = (const float*)d_in[4];
    const float* beta  = (const float*)d_in[5];
    float* out = (float*)d_out;

    float* rdenom = (float*)d_ws;                      // B*D*D f32 (1 MB)
    float* y      = rdenom + (size_t)BB * DD * DD;     // B*Q*O f32 (1 MB)

    gram_rdenom_kernel<<<dim3(DD / 16, DD / 16, BB), dim3(16, 16), 0, stream>>>(x, rdenom);
    base_kernel<<<dim3(BB * QQ), dim3(OO), 0, stream>>>(x, w1, bias, y);

    const size_t smem =
        (size_t)(64 * 258 + 16 * 256 + 8 * 2 * BUF_F) * sizeof(float);
    lc_gemm_kernel<<<dim3(QQ / 64, DD / 16, BB), dim3(256), smem, stream>>>(x, lcw, rdenom, y);

    bn_kernel<<<dim3(QQ), dim3(OO), 0, stream>>>(y, gamma, beta, out);
}
